// SignTransformerDeberta_7284264534253
// MI455X (gfx1250) — matrix operand order repaired and checked by reading, not compiled, not run on hardware
//
#include <hip/hip_runtime.h>
#include <math.h>

// ---------------------------------------------------------------------------
// CDNA5 (gfx1250) wave32 WMMA bf16 implementation of the SignTransformerDeberta
// forward pass.  All GEMMs lower to v_wmma_f32_16x16x32_bf16.
//
// Edge handling strategy: no per-element load masking anywhere.
//  - M/N edges: row/col indices are clamped into range at pointer setup;
//    out-of-range results are simply never stored (guarded epilogue).
//  - K is always a multiple of 32: ragged-K attention GEMMs read probability
//    matrices that we store with zero-padded columns (100->128, 80->96), and
//    the corresponding V over-reads land in explicitly zeroed pad slices.
// ---------------------------------------------------------------------------

typedef __attribute__((ext_vector_type(16))) __bf16 v16bf;
typedef __attribute__((ext_vector_type(8)))  float  v8f;

#define DEVFN static __device__ __forceinline__

DEVFN unsigned short f2bfu(float f) {           // f32 -> bf16 bits (RNE)
  unsigned u = __float_as_uint(f);
  u += 0x7FFFu + ((u >> 16) & 1u);
  return (unsigned short)(u >> 16);
}

DEVFN float wred_sum(float v) {                 // wave32 reduction
#pragma unroll
  for (int o = 16; o > 0; o >>= 1) v += __shfl_xor(v, o, 32);
  return v;
}

DEVFN int imin(int a, int b) { return a < b ? a : b; }

// ---------------------------------------------------------------------------
// Fused embedding:  LN(type_emb[id]*(id!=0)) ++ LN(lm_emb[id]*(id!=0)) ++ pos
// One wave per token (E=256 -> 8 elems/lane), writes f32 residual + bf16 copy.
// ---------------------------------------------------------------------------
__global__ __launch_bounds__(256) void embed_kernel(
    const int* __restrict__ tids, const int* __restrict__ lids,
    const float* __restrict__ xs, const float* __restrict__ ys,
    const float* __restrict__ zs,
    const float* __restrict__ temb, const float* __restrict__ lemb,
    const float* __restrict__ tg, const float* __restrict__ tb,
    const float* __restrict__ lg, const float* __restrict__ lb,
    const float* __restrict__ pw, const float* __restrict__ pb,
    float* __restrict__ fts, unsigned short* __restrict__ ftsB, int T)
{
  const int lane = threadIdx.x & 31;
  const int wid  = threadIdx.x >> 5;
  const long t = (long)blockIdx.x * 8 + wid;
  if (t >= T) return;
  float*          outF = fts  + t * 768;
  unsigned short* outB = ftsB + t * 768;

  for (int seg = 0; seg < 2; ++seg) {
    const int id = seg ? lids[t] : tids[t];
    const float* e  = (seg ? lemb : temb) + (long)id * 256;
    const float* g  = seg ? lg : tg;
    const float* bb = seg ? lb : tb;
    const float keep = (id != 0) ? 1.f : 0.f;
    float v[8], s = 0.f, s2 = 0.f;
#pragma unroll
    for (int j = 0; j < 8; ++j) {
      float w = e[lane + j * 32] * keep;
      v[j] = w; s += w; s2 += w * w;
    }
    s = wred_sum(s); s2 = wred_sum(s2);
    const float mean = s * (1.f / 256.f);
    const float var  = s2 * (1.f / 256.f) - mean * mean;
    const float rs   = rsqrtf(var + 1e-5f);
#pragma unroll
    for (int j = 0; j < 8; ++j) {
      const int ei = lane + j * 32;
      const float o = (v[j] - mean) * rs * g[ei] + bb[ei];
      outF[seg * 256 + ei] = o;
      outB[seg * 256 + ei] = f2bfu(o);
    }
  }
  const float x = xs[t], y = ys[t], z = zs[t];
#pragma unroll
  for (int j = 0; j < 8; ++j) {
    const int ei = lane + j * 32;
    const float o = pw[ei * 3 + 0] * x + pw[ei * 3 + 1] * y +
                    pw[ei * 3 + 2] * z + pb[ei];
    outF[512 + ei] = o;
    outB[512 + ei] = f2bfu(o);
  }
}

// ---------------------------------------------------------------------------
// Generic batched bf16 WMMA GEMM:  C = alpha*(A x op(B)) + bias, epilogue act.
//   A  : [M,K] bf16, row-major (lda mult of 8), per-batch stride sA
//   B  : bNK=1 -> weight [N,K] row-major (ldb mult of 8; C = A*B^T)
//        bNK=0 -> matrix [K,N] row-major
//   K must be a multiple of 32 (caller pads A columns with zeros if needed).
//   Cf : optional f32 out  (ldc, sC)     Cb: optional bf16 out (same ldc/sC)
// Wave tile 32x64 (2x4 WMMA frags), block tile 128x128, 8 waves/block.
// Clamped-row/col b128 streaming loads; no divergent load masking.
// Fragment layouts follow CDNA5 ISA 7.12.2 (wave32).
// ---------------------------------------------------------------------------
struct GemmParams {
  const unsigned short* A; long lda; long sA;
  const unsigned short* B; long ldb; long sB;
  const float* bias;
  float* Cf; unsigned short* Cb; long ldc; long sC;
  int M, N, K;
  float alpha;
  int bNK;     // 1: B is [N,K] ("W^T"), 0: B is [K,N]
  int act;     // 0 none, 1 relu, 2 exact gelu
};

union Frag { v16bf v; unsigned short u[16]; uint4 q[2]; };

__global__ __launch_bounds__(256) void wmma_gemm(GemmParams p)
{
  const int lane = threadIdx.x & 31;
  const int wid  = threadIdx.x >> 5;
  const int half = lane >> 4;       // 0/1
  const int l15  = lane & 15;
  const int rowBase = blockIdx.y * 128 + (wid >> 1) * 32;
  const int colBase = blockIdx.x * 128 + (wid & 1) * 64;
  const long bz = blockIdx.z;
  const unsigned short* A = p.A + bz * p.sA;
  const unsigned short* B = p.B + bz * p.sB;

  v8f acc[2][4];
#pragma unroll
  for (int i = 0; i < 2; ++i)
#pragma unroll
    for (int j = 0; j < 4; ++j) { v8f z = {}; acc[i][j] = z; }

  Frag a[2], b[4];

  // per-lane streaming pointers; rows/cols clamped into range (results for
  // clamped lanes are discarded by the guarded epilogue).
  const unsigned short* pa[2];
#pragma unroll
  for (int mt = 0; mt < 2; ++mt)
    pa[mt] = A + (long)imin(rowBase + mt * 16 + l15, p.M - 1) * p.lda + half * 8;

  if (p.bNK) {
    const unsigned short* pb[4];
#pragma unroll
    for (int nt = 0; nt < 4; ++nt)
      pb[nt] = B + (long)imin(colBase + nt * 16 + l15, p.N - 1) * p.ldb + half * 8;
    for (int k = 0; k < p.K; k += 32) {
#pragma unroll
      for (int mt = 0; mt < 2; ++mt) {
        a[mt].q[0] = *(const uint4*)(pa[mt]);
        a[mt].q[1] = *(const uint4*)(pa[mt] + 16);
        pa[mt] += 32;
      }
#pragma unroll
      for (int nt = 0; nt < 4; ++nt) {
        b[nt].q[0] = *(const uint4*)(pb[nt]);
        b[nt].q[1] = *(const uint4*)(pb[nt] + 16);
        pb[nt] += 32;
      }
#pragma unroll
      for (int mt = 0; mt < 2; ++mt)
#pragma unroll
        for (int nt = 0; nt < 4; ++nt)
          acc[mt][nt] = __builtin_amdgcn_wmma_f32_16x16x32_bf16(
              false, a[mt].v, false, b[nt].v, (short)0, acc[mt][nt], false, false);
    }
  } else {
    const unsigned short* pb[4];
#pragma unroll
    for (int nt = 0; nt < 4; ++nt)
      pb[nt] = B + (long)(half * 8) * p.ldb + imin(colBase + nt * 16 + l15, p.N - 1);
    for (int k = 0; k < p.K; k += 32) {
#pragma unroll
      for (int mt = 0; mt < 2; ++mt) {
        a[mt].q[0] = *(const uint4*)(pa[mt]);
        a[mt].q[1] = *(const uint4*)(pa[mt] + 16);
        pa[mt] += 32;
      }
#pragma unroll
      for (int nt = 0; nt < 4; ++nt) {
        const unsigned short* s = pb[nt];
#pragma unroll
        for (int j = 0; j < 16; ++j) { b[nt].u[j] = s[(long)(j + (j & 8)) * p.ldb]; }
        pb[nt] += 32 * p.ldb;
      }
#pragma unroll
      for (int mt = 0; mt < 2; ++mt)
#pragma unroll
        for (int nt = 0; nt < 4; ++nt)
          acc[mt][nt] = __builtin_amdgcn_wmma_f32_16x16x32_bf16(
              false, a[mt].v, false, b[nt].v, (short)0, acc[mt][nt], false, false);
    }
  }

  // epilogue (C/D layout: lane -> col n=l15, rows m = frag + r + 8*half)
#pragma unroll
  for (int nt = 0; nt < 4; ++nt) {
    const int n = colBase + nt * 16 + l15;
    if (n >= p.N) continue;
    const float bias = p.bias ? p.bias[n] : 0.f;
#pragma unroll
    for (int mt = 0; mt < 2; ++mt) {
#pragma unroll
      for (int r = 0; r < 8; ++r) {
        const int m = rowBase + mt * 16 + half * 8 + r;
        if (m >= p.M) continue;
        float v = acc[mt][nt][r] * p.alpha + bias;
        if (p.act == 1)      v = fmaxf(v, 0.f);
        else if (p.act == 2) v = 0.5f * v * (1.f + erff(v * 0.70710678118654752f));
        const long idx = (long)m * p.ldc + n;
        if (p.Cf) p.Cf[bz * p.sC + idx] = v;
        if (p.Cb) p.Cb[bz * p.sC + idx] = f2bfu(v);
      }
    }
  }
}

// ---------------------------------------------------------------------------
// Row LayerNorm:  out = LN(x[row] (+ resid[row])) * g + b  (f32 and/or bf16)
// ---------------------------------------------------------------------------
__global__ __launch_bounds__(256) void ln_rows(
    const float* __restrict__ x, const float* __restrict__ resid,
    const float* __restrict__ g, const float* __restrict__ b,
    float eps, int D, float* __restrict__ outF, unsigned short* __restrict__ outB)
{
  const long row = blockIdx.x;
  const int lane = threadIdx.x & 31;
  const int wid  = threadIdx.x >> 5;
  __shared__ float redA[8], redB[8];
  float s = 0.f, s2 = 0.f;
  for (int d = threadIdx.x; d < D; d += 256) {
    float v = x[row * D + d] + (resid ? resid[row * D + d] : 0.f);
    s += v; s2 += v * v;
  }
  s = wred_sum(s); s2 = wred_sum(s2);
  if (lane == 0) { redA[wid] = s; redB[wid] = s2; }
  __syncthreads();
  if (threadIdx.x == 0) {
    float a = 0.f, c = 0.f;
    for (int i = 0; i < 8; ++i) { a += redA[i]; c += redB[i]; }
    const float mean = a / D;
    const float var  = c / D - mean * mean;
    redA[0] = mean; redB[0] = rsqrtf(var + eps);
  }
  __syncthreads();
  const float mean = redA[0], rs = redB[0];
  for (int d = threadIdx.x; d < D; d += 256) {
    float v = x[row * D + d] + (resid ? resid[row * D + d] : 0.f);
    float o = (v - mean) * rs * g[d] + b[d];
    if (outF) outF[row * D + d] = o;
    if (outB) outB[row * D + d] = f2bfu(o);
  }
}

// mean over middle axis:  out[row,d] = mean_i in[(row*n+i), d]
__global__ __launch_bounds__(256) void mean_over(
    const float* __restrict__ in, float* __restrict__ outF,
    unsigned short* __restrict__ outB, int n, int D)
{
  const long row = blockIdx.x;
  for (int d = threadIdx.x; d < D; d += 256) {
    float s = 0.f;
    for (int i = 0; i < n; ++i) s += in[(row * n + i) * (long)D + d];
    const float o = s / (float)n;
    outF[row * D + d] = o;
    if (outB) outB[row * D + d] = f2bfu(o);
  }
}

// row softmax (n <= 128), bf16 out; columns [n, ldo) are written as zeros so
// downstream GEMMs can run with K = ldo (multiple of 32).
__global__ __launch_bounds__(128) void softmax_rows(
    const float* __restrict__ s, unsigned short* __restrict__ p, int n, int ldo)
{
  const int t = threadIdx.x;
  __shared__ float sh[128];
  float v = (t < n) ? s[(long)blockIdx.x * n + t] : -3.4028235e38f;
  sh[t] = v; __syncthreads();
#pragma unroll
  for (int o = 64; o > 0; o >>= 1) { if (t < o) sh[t] = fmaxf(sh[t], sh[t + o]); __syncthreads(); }
  const float mx = sh[0]; __syncthreads();
  float e = (t < n) ? expf(v - mx) : 0.f;
  sh[t] = e; __syncthreads();
#pragma unroll
  for (int o = 64; o > 0; o >>= 1) { if (t < o) sh[t] += sh[t + o]; __syncthreads(); }
  const float inv = 1.f / sh[0];
  if (t < n)        p[(long)blockIdx.x * ldo + t] = f2bfu(e * inv);
  else if (t < ldo) p[(long)blockIdx.x * ldo + t] = 0;   // zero pad columns
}

// DeBERTa: combine qk + disentangled terms, XSoftmax with mask, bf16 probs.
// For F=80 < ATT_SPAN/2 the log bucket is the identity: idx = q-k+256.
__global__ __launch_bounds__(128) void deberta_softmax(
    const float* __restrict__ qk, const float* __restrict__ c2p,
    const float* __restrict__ p2c, const int* __restrict__ mask,
    unsigned short* __restrict__ probs, int Fn, int Ln, int ldo)
{
  const int b = blockIdx.x / Fn, q = blockIdx.x % Fn;
  const int t = threadIdx.x;                       // key index
  __shared__ float sh[128];
  const bool mq = mask[((long)b * Fn + q) * Ln] > 0;
  float v = -3.4028235e38f; bool valid = false;
  if (t < Fn) {
    const bool mk = mask[((long)b * Fn + t) * Ln] > 0;
    valid = mq && mk;
    const int idx = q - t + 256;                   // in [176,336) for F=80
    const float sc = qk[((long)b * Fn + q) * Fn + t]
                   + c2p[((long)b * Fn + q) * 512 + idx]
                   + p2c[((long)b * Fn + t) * 512 + idx];
    v = valid ? sc : -3.4028235e38f;
  }
  sh[t] = v; __syncthreads();
#pragma unroll
  for (int o = 64; o > 0; o >>= 1) { if (t < o) sh[t] = fmaxf(sh[t], sh[t + o]); __syncthreads(); }
  const float mx = sh[0]; __syncthreads();
  float e = (t < Fn) ? expf(v - mx) : 0.f;
  sh[t] = e; __syncthreads();
#pragma unroll
  for (int o = 64; o > 0; o >>= 1) { if (t < o) sh[t] += sh[t + o]; __syncthreads(); }
  const float inv = 1.f / sh[0];
  if (t < Fn)       probs[((long)b * Fn + q) * ldo + t] = f2bfu(valid ? e * inv : 0.f);
  else if (t < ldo) probs[((long)b * Fn + q) * ldo + t] = 0;   // zero pad columns
}

// classifier head: out[b,c] = pooled[b,:] . head_w[c,:] + head_b[c]
__global__ __launch_bounds__(256) void head_kernel(
    const float* __restrict__ pooled, const float* __restrict__ w,
    const float* __restrict__ bias, float* __restrict__ out, int D, int NC)
{
  const int b = blockIdx.x, c = threadIdx.x;
  if (c >= NC) return;
  float s = bias[c];
  for (int d = 0; d < D; ++d) s += pooled[(long)b * D + d] * w[(long)c * D + d];
  out[(long)b * NC + c] = s;
}

__global__ __launch_bounds__(256) void cvt_bf16(
    const float* __restrict__ in, unsigned short* __restrict__ out, long n)
{
  long i = (long)blockIdx.x * 256 + threadIdx.x;
  const long stride = (long)gridDim.x * 256;
  for (; i < n; i += stride) out[i] = f2bfu(in[i]);
}

__global__ __launch_bounds__(256) void zero_fill(
    unsigned short* __restrict__ p, long n)
{
  long i = (long)blockIdx.x * 256 + threadIdx.x;
  const long stride = (long)gridDim.x * 256;
  for (; i < n; i += stride) p[i] = 0;
}

// ---------------------------------------------------------------------------
// Host launch sequence
// ---------------------------------------------------------------------------
static void launch_gemm(hipStream_t st,
    const unsigned short* A, long lda, long sA,
    const unsigned short* B, long ldb, long sB, int bNK,
    const float* bias, float* Cf, unsigned short* Cb, long ldc, long sC,
    int M, int N, int K, int batch, float alpha, int act)
{
  GemmParams p;
  p.A = A; p.lda = lda; p.sA = sA;
  p.B = B; p.ldb = ldb; p.sB = sB;
  p.bias = bias; p.Cf = Cf; p.Cb = Cb; p.ldc = ldc; p.sC = sC;
  p.M = M; p.N = N; p.K = K; p.alpha = alpha; p.bNK = bNK; p.act = act;
  dim3 grid((N + 127) / 128, (M + 127) / 128, batch);
  wmma_gemm<<<grid, dim3(256), 0, st>>>(p);
}

extern "C" void kernel_launch(void* const* d_in, const int* in_sizes, int n_in,
                              void* d_out, int out_size, void* d_ws, size_t ws_size,
                              hipStream_t stream)
{
  (void)in_sizes; (void)n_in; (void)out_size; (void)ws_size;
  enum { Bn = 8, Fn = 80, Ln = 100, D = 768, DF = 3072, NC = 250, R = 512 };
  enum { PLn = 128, PFn = 96 };     // padded prob strides == padded K (mult 32)
  const int  S = Bn * Fn;           // 640
  const long T = (long)S * Ln;      // 64000

  // ---- inputs (setup_inputs order) ----
  const int*   type_ids = (const int*)d_in[0];
  const int*   lm_ids   = (const int*)d_in[1];
  const float* xs = (const float*)d_in[2];
  const float* ys = (const float*)d_in[3];
  const float* zs = (const float*)d_in[4];
  const int*   mask = (const int*)d_in[5];
  const float* type_emb = (const float*)d_in[6];
  const float* lm_emb   = (const float*)d_in[7];
  const float* type_g = (const float*)d_in[8];  const float* type_b = (const float*)d_in[9];
  const float* lmg    = (const float*)d_in[10]; const float* lmb    = (const float*)d_in[11];
  const float* pos_w  = (const float*)d_in[12]; const float* pos_b  = (const float*)d_in[13];
  const float* lt_in_w  = (const float*)d_in[14]; const float* lt_in_b  = (const float*)d_in[15];
  const float* lt_out_w = (const float*)d_in[16]; const float* lt_out_b = (const float*)d_in[17];
  const float* ln1_g = (const float*)d_in[18]; const float* ln1_b = (const float*)d_in[19];
  const float* ff1_w = (const float*)d_in[20]; const float* ff1_b = (const float*)d_in[21];
  const float* ff2_w = (const float*)d_in[22]; const float* ff2_b = (const float*)d_in[23];
  const float* ln2_g = (const float*)d_in[24]; const float* ln2_b = (const float*)d_in[25];
  const float* rel_emb = (const float*)d_in[26];
  const float* rel_g = (const float*)d_in[27]; const float* rel_b = (const float*)d_in[28];
  const float* dq_w = (const float*)d_in[29]; const float* dq_b = (const float*)d_in[30];
  const float* dk_w = (const float*)d_in[31]; const float* dk_b = (const float*)d_in[32];
  const float* dv_w = (const float*)d_in[33]; const float* dv_b = (const float*)d_in[34];
  const float* dao_w = (const float*)d_in[35]; const float* dao_b = (const float*)d_in[36];
  const float* da_g = (const float*)d_in[37]; const float* da_b = (const float*)d_in[38];
  const float* di_w = (const float*)d_in[39]; const float* di_b = (const float*)d_in[40];
  const float* do_w = (const float*)d_in[41]; const float* do_b = (const float*)d_in[42];
  const float* do_g = (const float*)d_in[43]; const float* dob = (const float*)d_in[44];
  const float* head_w = (const float*)d_in[45]; const float* head_b = (const float*)d_in[46];

  // ---- workspace layout (256B aligned slices) ----
  size_t off = 0;
  char* base = (char*)d_ws;
  auto give = [&](size_t bytes) -> char* {
    char* p = base + off;
    off += (bytes + 255) & ~(size_t)255;
    return p;
  };
  float*          fts   = (float*)give(T * D * 4);        // f32 residual stream
  char*           qkvR  = give(T * 3 * D * 2);            // qkv bf16 region
  unsigned short* qkv_b = (unsigned short*)qkvR;
  float*          tmp   = (float*)qkvR;                   // alias (dead q,k)
  unsigned short* ffm_b = (unsigned short*)(qkvR + T * D * 4); // alias (dead v)
  unsigned short* qkvPad = (unsigned short*)give(32L * 3 * D * 2); // zeroed: V pad rows
  unsigned short* h_b   = (unsigned short*)give(T * D * 2);
  float*          scrs  = (float*)give((long)S * Ln * Ln * 4);
  unsigned short* prb_b = (unsigned short*)give((long)S * Ln * PLn * 2);
  unsigned short* ctx_b = (unsigned short*)give(T * D * 2);
  float*          hF    = (float*)give((long)S * D * 4);
  unsigned short* hF_b  = (unsigned short*)give((long)S * D * 2);
  unsigned short* rln_b = (unsigned short*)give((long)R * D * 2);
  unsigned short* pq_b  = (unsigned short*)give((long)R * D * 2);
  unsigned short* pk_b  = (unsigned short*)give((long)R * D * 2);
  unsigned short* qF_b  = (unsigned short*)give((long)S * D * 2);
  unsigned short* kF_b  = (unsigned short*)give((long)S * D * 2);
  unsigned short* vF_b  = (unsigned short*)give((long)(S + 16) * D * 2); // +16 zero rows
  float*          c2p   = (float*)give((long)Bn * Fn * R * 4);
  float*          p2c   = (float*)give((long)Bn * Fn * R * 4);
  float*          dqk   = (float*)give((long)Bn * Fn * Fn * 4);
  unsigned short* dpr_b = (unsigned short*)give((long)Bn * Fn * PFn * 2);
  unsigned short* dcx_b = (unsigned short*)give((long)S * D * 2);
  unsigned short* itm_b = (unsigned short*)give((long)S * DF * 2);
  float*          pooled = (float*)give((long)Bn * D * 4);
  unsigned short* w_ltin = (unsigned short*)give(3L * D * D * 2);
  unsigned short* w_ltout = (unsigned short*)give((long)D * D * 2);
  unsigned short* w_ff1 = (unsigned short*)give((long)D * D * 2);
  unsigned short* w_ff2 = (unsigned short*)give((long)D * D * 2);
  unsigned short* w_dq  = (unsigned short*)give((long)D * D * 2);
  unsigned short* w_dk  = (unsigned short*)give((long)D * D * 2);
  unsigned short* w_dv  = (unsigned short*)give((long)D * D * 2);
  unsigned short* w_dao = (unsigned short*)give((long)D * D * 2);
  unsigned short* w_di  = (unsigned short*)give((long)DF * D * 2);
  unsigned short* w_do  = (unsigned short*)give((long)D * DF * 2);

  auto cvt = [&](const float* in, unsigned short* out, long n) {
    int g = (int)((n + 255) / 256); if (g > 4096) g = 4096;
    cvt_bf16<<<g, 256, 0, stream>>>(in, out, n);
  };
  // ---- zero pads (finite 0s for padded-K over-reads) ----
  zero_fill<<<64, 256, 0, stream>>>(qkvPad, 32L * 3 * D);
  zero_fill<<<48, 256, 0, stream>>>(vF_b + (long)S * D, 16L * D);
  // ---- weight conversions to bf16 ----
  cvt(lt_in_w,  w_ltin, 3L * D * D);
  cvt(lt_out_w, w_ltout, (long)D * D);
  cvt(ff1_w, w_ff1, (long)D * D);
  cvt(ff2_w, w_ff2, (long)D * D);
  cvt(dq_w,  w_dq,  (long)D * D);
  cvt(dk_w,  w_dk,  (long)D * D);
  cvt(dv_w,  w_dv,  (long)D * D);
  cvt(dao_w, w_dao, (long)D * D);
  cvt(di_w,  w_di,  (long)DF * D);
  cvt(do_w,  w_do,  (long)D * DF);

  const float invSqrtD = 0.03608439182435161f;   // 1/sqrt(768)
  const float inv48    = 1.f / 48.f;             // 1/sqrt(768*3)

  // ---- embeddings + LN + pos proj ----
  embed_kernel<<<(unsigned)((T + 7) / 8), 256, 0, stream>>>(
      type_ids, lm_ids, xs, ys, zs, type_emb, lm_emb,
      type_g, type_b, lmg, lmb, pos_w, pos_b, fts, h_b, (int)T);

  // ---- landmark transformer layer ----
  // qkv = h @ lt_in_w^T + b         [T,2304] bf16
  launch_gemm(stream, h_b, D, 0, w_ltin, D, 0, 1, lt_in_b,
              nullptr, qkv_b, 3 * D, 0, (int)T, 3 * D, D, 1, 1.f, 0);
  // scores = q k^T / sqrt(D)        batched [100,100]
  launch_gemm(stream, qkv_b, 3 * D, (long)Ln * 3 * D,
              qkv_b + D, 3 * D, (long)Ln * 3 * D, 1, nullptr,
              scrs, nullptr, Ln, (long)Ln * Ln, Ln, Ln, D, S, invSqrtD, 0);
  softmax_rows<<<S * Ln, 128, 0, stream>>>(scrs, prb_b, Ln, PLn);
  // ctx = probs @ v  (K padded 100->128; pad cols zero, V pad rows land in
  //  the next sequence's finite data or the zeroed qkvPad slice)
  launch_gemm(stream, prb_b, PLn, (long)Ln * PLn,
              qkv_b + 2 * D, 3 * D, (long)Ln * 3 * D, 0, nullptr,
              nullptr, ctx_b, D, (long)Ln * D, Ln, D, PLn, S, 1.f, 0);
  // attn = ctx @ lt_out_w^T + b  -> tmp (f32, aliases dead q,k)
  launch_gemm(stream, ctx_b, D, 0, w_ltout, D, 0, 1, lt_out_b,
              tmp, nullptr, D, 0, (int)T, D, D, 1, 1.f, 0);
  ln_rows<<<(unsigned)T, 256, 0, stream>>>(tmp, fts, ln1_g, ln1_b, 1e-5f, D, fts, h_b);
  // ff: relu(h W1^T + b1) W2^T + b2
  launch_gemm(stream, h_b, D, 0, w_ff1, D, 0, 1, ff1_b,
              nullptr, ffm_b, D, 0, (int)T, D, D, 1, 1.f, 1);
  launch_gemm(stream, ffm_b, D, 0, w_ff2, D, 0, 1, ff2_b,
              tmp, nullptr, D, 0, (int)T, D, D, 1, 1.f, 0);
  ln_rows<<<(unsigned)T, 256, 0, stream>>>(tmp, fts, ln2_g, ln2_b, 1e-5f, D, fts, nullptr);
  // mean over landmarks -> [S,768]
  mean_over<<<S, 256, 0, stream>>>(fts, hF, hF_b, Ln, D);

  // ---- DeBERTa layer over frames ----
  ln_rows<<<R, 256, 0, stream>>>(rel_emb, nullptr, rel_g, rel_b, 1e-7f, D, nullptr, rln_b);
  launch_gemm(stream, rln_b, D, 0, w_dq, D, 0, 1, dq_b,
              nullptr, pq_b, D, 0, R, D, D, 1, 1.f, 0);
  launch_gemm(stream, rln_b, D, 0, w_dk, D, 0, 1, dk_b,
              nullptr, pk_b, D, 0, R, D, D, 1, 1.f, 0);
  launch_gemm(stream, hF_b, D, 0, w_dq, D, 0, 1, dq_b,
              nullptr, qF_b, D, 0, S, D, D, 1, 1.f, 0);
  launch_gemm(stream, hF_b, D, 0, w_dk, D, 0, 1, dk_b,
              nullptr, kF_b, D, 0, S, D, D, 1, 1.f, 0);
  launch_gemm(stream, hF_b, D, 0, w_dv, D, 0, 1, dv_b,
              nullptr, vF_b, D, 0, S, D, D, 1, 1.f, 0);
  // qk / 48                      batched [80,80]
  launch_gemm(stream, qF_b, D, (long)Fn * D, kF_b, D, (long)Fn * D, 1, nullptr,
              dqk, nullptr, Fn, (long)Fn * Fn, Fn, Fn, D, Bn, inv48, 0);
  // c2p = q posK^T / 48          batched [80,512], shared B
  launch_gemm(stream, qF_b, D, (long)Fn * D, pk_b, D, 0, 1, nullptr,
              c2p, nullptr, R, (long)Fn * R, Fn, R, D, Bn, inv48, 0);
  // p2c = k posQ^T / 48
  launch_gemm(stream, kF_b, D, (long)Fn * D, pq_b, D, 0, 1, nullptr,
              p2c, nullptr, R, (long)Fn * R, Fn, R, D, Bn, inv48, 0);
  deberta_softmax<<<Bn * Fn, 128, 0, stream>>>(dqk, c2p, p2c, mask, dpr_b, Fn, Ln, PFn);
  // ctx = probs @ v  (K padded 80->96; V pad rows land in next batch rows or
  //  the 16 zeroed rows appended to vF)
  launch_gemm(stream, dpr_b, PFn, (long)Fn * PFn, vF_b, D, (long)Fn * D, 0, nullptr,
              nullptr, dcx_b, D, (long)Fn * D, Fn, D, PFn, Bn, 1.f, 0);
  launch_gemm(stream, dcx_b, D, 0, w_dao, D, 0, 1, dao_b,
              tmp, nullptr, D, 0, S, D, D, 1, 1.f, 0);
  ln_rows<<<S, 256, 0, stream>>>(tmp, hF, da_g, da_b, 1e-7f, D, hF, hF_b);
  // intermediate = gelu(h Wi^T + bi)  [640,3072]
  launch_gemm(stream, hF_b, D, 0, w_di, D, 0, 1, di_b,
              nullptr, itm_b, DF, 0, S, DF, D, 1, 1.f, 2);
  launch_gemm(stream, itm_b, DF, 0, w_do, DF, 0, 1, do_b,
              tmp, nullptr, D, 0, S, D, DF, 1, 1.f, 0);
  ln_rows<<<S, 256, 0, stream>>>(tmp, hF, do_g, dob, 1e-7f, D, hF, nullptr);

  // ---- pool + head ----
  mean_over<<<Bn, 256, 0, stream>>>(hF, pooled, nullptr, Fn, D);
  head_kernel<<<Bn, 256, 0, stream>>>(pooled, head_w, head_b, (float*)d_out, D, NC);
}
